// DistanceTransformerLayer_1881195676265
// MI455X (gfx1250) — matrix operand order repaired and checked by reading, not compiled, not run on hardware
//
#include <hip/hip_runtime.h>

// ---------------------------------------------------------------------------
// DistanceTransformerLayer for MI455X (gfx1250), wave32 + WMMA.
//
// Dominant cost: de = relu(dist*W1+b1) @ W2 over all B*N*N pairs
//   = 73728 x 1024 x 256 GEMM (~38.7 GFLOP) -> v_wmma_f32_16x16x32_bf16.
// Fused flash-style: de-GEMM + decay/q/k combine + per-channel online softmax
// over j + sum_j(p*v); the [B,N,N,256] tensors (75 MB each) never hit HBM.
// v3: all weights pre-converted ONCE to bf16 and pre-transposed to [N][K];
// B fragments are loaded straight from global (global_load_b128, L2/WGP$-hot),
// so the hot loops only stage/convert the A side. Double-buffered A slices,
// one barrier per K-step, global_prefetch_b8 on upcoming W1/W2T rows.
// ---------------------------------------------------------------------------

#define NB 2
#define NTOK 192
#define DIMC 256
#define HID 1024
#define QKVLD 768
#define SCALING 0.17677669529663687f   // 32^-0.5

typedef __attribute__((ext_vector_type(16))) __bf16 v16bf;
typedef __attribute__((ext_vector_type(8)))  float  v8f;

static __device__ __forceinline__ __bf16 tobf(float f) { return (__bf16)f; }

// A-fragment (16x32 bf16, M = lane&15): VGPR0..3 = K {0..7}+off, VGPR4..7 = K {16..23}+off, off=(lane&16)?8:0
static __device__ __forceinline__ v16bf frag_a(const __bf16* row, unsigned lane) {
    unsigned off = (lane & 16u) ? 8u : 0u;
    union { uint4 u[2]; v16bf v; } x;
    x.u[0] = *(const uint4*)(row + off);
    x.u[1] = *(const uint4*)(row + 16 + off);
    return x.v;
}

// B-fragment (32x16 bf16, N = lane&15): VGPR0..3 = K {0..7}+off, VGPR4..7 = K {16..23}+off, off=(lane&16)?8:0
// `row` = bf16 row of B^T (one output channel, K-contiguous) + k0
static __device__ __forceinline__ v16bf frag_b(const __bf16* row, unsigned lane) {
    unsigned off = (lane & 16u) ? 8u : 0u;
    union { uint4 u[2]; v16bf v; } x;
    x.u[0] = *(const uint4*)(row + off);
    x.u[1] = *(const uint4*)(row + 16 + off);
    return x.v;
}

static __device__ __forceinline__ v8f wmma_bf16(v16bf a, v16bf b, v8f c) {
    return __builtin_amdgcn_wmma_f32_16x16x32_bf16(false, a, false, b, (short)0, c, false, false);
}

// ---------------------------------------------------------------------------
// Setup: W[K,N] fp32 -> WT[N,K] bf16 (once per launch; ~1M elems total)
// ---------------------------------------------------------------------------
__global__ __launch_bounds__(256)
void cvt_transpose_kernel(const float* __restrict__ W, __bf16* __restrict__ WT,
                          int K, int N) {
    int idx = blockIdx.x * 256 + threadIdx.x;
    if (idx >= K * N) return;
    int k = idx / N, n = idx - k * N;          // reads coalesced over n
    WT[(size_t)n * K + k] = tobf(W[idx]);
}

// ---------------------------------------------------------------------------
// Generic GEMM: C[M,Ncols] = act(A[M,K] @ B + bias) (+ residual),
// B given as BT[Ncols][K] bf16 in global (fragments loaded directly).
// block = 256 threads (8 waves), tile 128(M) x 32(N), double-buffered A.
// ---------------------------------------------------------------------------
static __device__ __forceinline__ void gemm_fill_a(
    const float* __restrict__ A, int K, int m0, int k0, __bf16* Adst, int t) {
    int row = t >> 1, half = t & 1;
    const float* ap = A + (size_t)(m0 + row) * K + k0 + half * 16;
    __bf16* dst = Adst + row * 32 + half * 16;
#pragma unroll
    for (int u = 0; u < 16; ++u) dst[u] = tobf(ap[u]);
}

__global__ __launch_bounds__(256)
void gemm_bf16_kernel(const float* __restrict__ A, const __bf16* __restrict__ BT,
                      const float* __restrict__ bias, const float* __restrict__ Res,
                      float* __restrict__ C, int M, int K, int Ncols, int do_relu) {
    __shared__ __bf16 As[2][128 * 32];   // 2 x 8 KB
    const int t = threadIdx.x;
    const unsigned lane = t & 31;
    const int w = t >> 5;
    const int m0 = blockIdx.y * 128;
    const int n0 = blockIdx.x * 32;

    const __bf16* brow0 = BT + (size_t)(n0 + (lane & 15)) * K;
    const __bf16* brow1 = BT + (size_t)(n0 + 16 + (lane & 15)) * K;

    v8f acc[2];
    for (int i = 0; i < 2; ++i)
        for (int e = 0; e < 8; ++e) acc[i][e] = 0.0f;

    gemm_fill_a(A, K, m0, 0, As[0], t);
    for (int k0 = 0; k0 < K; k0 += 32) {
        const int buf = (k0 >> 5) & 1;
        __syncthreads();
        if (k0 + 32 < K) gemm_fill_a(A, K, m0, k0 + 32, As[buf ^ 1], t);
        v16bf af  = frag_a(&As[buf][(w * 16 + (lane & 15)) * 32], lane);
        v16bf bf0 = frag_b(brow0 + k0, lane);
        v16bf bf1 = frag_b(brow1 + k0, lane);
        acc[0] = wmma_bf16(af, bf0, acc[0]);
        acc[1] = wmma_bf16(af, bf1, acc[1]);
    }

#pragma unroll
    for (int ct = 0; ct < 2; ++ct) {
#pragma unroll
        for (int r = 0; r < 8; ++r) {
            int m = m0 + w * 16 + r + 8 * (lane >> 4);
            int n = n0 + ct * 16 + (lane & 15);
            float c = acc[ct][r] + bias[n];
            if (do_relu) c = fmaxf(c, 0.0f);
            if (Res) c += Res[(size_t)m * Ncols + n];
            C[(size_t)m * Ncols + n] = c;
        }
    }
}

// ---------------------------------------------------------------------------
// Fused distance-attention.
// grid = (DIM/64 channel groups, N/16 i-tiles, B); block = 512 (16 waves).
// Wave w owns query row i = itile*16+w and 64 channels (4 C tiles).
// W2T[256][1024] bf16: B fragments straight from global.
// ---------------------------------------------------------------------------
static __device__ __forceinline__ void attn_fill_a(
    const float* __restrict__ W1, const float* __restrict__ b1,
    const __bf16* __restrict__ W2T, const float* __restrict__ distS,
    __bf16* Adst, int K0, int d0, int t) {
    {   // A: hidden[p][h] = relu(dist_p * W1[h] + b1[h]) -> bf16, 16/thread
        int p = t >> 1, half = t & 1;
        float dp = distS[p];
        const float* w1p = W1 + K0 + half * 16;
        const float* b1p = b1 + K0 + half * 16;
        __bf16* dst = Adst + p * 32 + half * 16;
#pragma unroll
        for (int u = 0; u < 16; ++u)
            dst[u] = tobf(fmaxf(dp * w1p[u] + b1p[u], 0.0f));
    }
    // pull next K-slice of W2T / W1 toward the WGP (global_prefetch_b8)
    if (t < 64) {
        __builtin_prefetch(&W2T[(size_t)(d0 + t) * HID + K0 + 32], 0, 1);
        if (t == 0) {
            __builtin_prefetch(&W1[K0 + 32], 0, 1);
            __builtin_prefetch(&b1[K0 + 32], 0, 1);
        }
    }
}

__global__ __launch_bounds__(512)
void attn_kernel(const float* __restrict__ qkv, const float* __restrict__ pos,
                 const unsigned char* __restrict__ mask,
                 const float* __restrict__ W1, const float* __restrict__ b1,
                 const __bf16* __restrict__ W2T, const float* __restrict__ b2,
                 float* __restrict__ attn_out) {
    __shared__ __bf16 As[2][256 * 32];   // activations, 2 x 16 KB
    __shared__ float distS[256];
    __shared__ float decayS[256];
    __shared__ float qS[16 * 64];
    __shared__ float vS[16 * 64];
    __shared__ unsigned char mjS[16];

    const int t = threadIdx.x;
    const unsigned lane = t & 31;
    const int w = t >> 5;               // wave id = local query row
    const int dg = blockIdx.x;          // channel group (64 channels)
    const int itile = blockIdx.y;
    const int b = blockIdx.z;
    const int i = itile * 16 + w;
    const int dl = lane & 15;
    const int d0 = dg * 64;

    const __bf16* brow[4];
#pragma unroll
    for (int ct = 0; ct < 4; ++ct)
        brow[ct] = W2T + (size_t)(d0 + ct * 16 + dl) * HID;

    float kv[4], b2v[4];
#pragma unroll
    for (int ct = 0; ct < 4; ++ct) {
        int d = d0 + ct * 16 + dl;
        kv[ct]  = qkv[(size_t)(b * NTOK + i) * QKVLD + DIMC + d];  // k_i
        b2v[ct] = b2[d];
    }
    const int mask_i = mask[b * NTOK + i];

    float m_[4] = {-3.0e38f, -3.0e38f, -3.0e38f, -3.0e38f};
    float s_[4] = {0.0f, 0.0f, 0.0f, 0.0f};
    float o_[4] = {0.0f, 0.0f, 0.0f, 0.0f};

    for (int jt = 0; jt < NTOK / 16; ++jt) {
        __syncthreads();   // previous epilogue done with decayS/qS/vS/mjS
        if (t < 256) {     // dist / decay for the 16x16 (i,j) tile
            int il = t >> 4, jl = t & 15;
            int gi = itile * 16 + il, gj = jt * 16 + jl;
            float dx = pos[(size_t)(b * NTOK + gi) * 3 + 0] - pos[(size_t)(b * NTOK + gj) * 3 + 0];
            float dy = pos[(size_t)(b * NTOK + gi) * 3 + 1] - pos[(size_t)(b * NTOK + gj) * 3 + 1];
            float dz = pos[(size_t)(b * NTOK + gi) * 3 + 2] - pos[(size_t)(b * NTOK + gj) * 3 + 2];
            float sq = dx * dx + dy * dy + dz * dz;
            distS[t]  = (sq > 0.0f) ? sqrtf(sq) : 0.0f;
            decayS[t] = __expf(-sq * 0.1f);
        }
        {   // q_j / v_j tiles for this channel group, 2 each
            int jl = t >> 5, dd = t & 31;
            int gj = jt * 16 + jl;
            const float* qr = qkv + (size_t)(b * NTOK + gj) * QKVLD;
            qS[jl * 64 + dd]      = qr[d0 + dd];
            qS[jl * 64 + dd + 32] = qr[d0 + dd + 32];
            vS[jl * 64 + dd]      = qr[2 * DIMC + d0 + dd];
            vS[jl * 64 + dd + 32] = qr[2 * DIMC + d0 + dd + 32];
        }
        if (t < 16) mjS[t] = mask[b * NTOK + jt * 16 + t];
        __syncthreads();   // distS visible before first A-fill

        v8f acc[4];
        for (int c = 0; c < 4; ++c)
            for (int e = 0; e < 8; ++e) acc[c][e] = 0.0f;

        attn_fill_a(W1, b1, W2T, distS, As[0], 0, d0, t);
        for (int ks = 0; ks < HID / 32; ++ks) {
            const int buf = ks & 1;
            const int K0 = ks * 32;
            __syncthreads();
            if (ks + 1 < HID / 32)
                attn_fill_a(W1, b1, W2T, distS, As[buf ^ 1], K0 + 32, d0, t);
            v16bf af = frag_a(&As[buf][(w * 16 + dl) * 32], lane);
#pragma unroll
            for (int ct = 0; ct < 4; ++ct) {
                v16bf bf = frag_b(brow[ct] + K0, lane);
                acc[ct] = wmma_bf16(af, bf, acc[ct]);
            }
        }

        // epilogue: combine + per-channel online softmax over j + p*v accumulate
#pragma unroll
        for (int ct = 0; ct < 4; ++ct) {
            float wv[8];
            float tm = -3.0e38f;
#pragma unroll
            for (int r = 0; r < 8; ++r) {
                int jl = r + ((lane >> 4) << 3);
                float de = acc[ct][r] + b2v[ct];
                float x = decayS[w * 16 + jl] * (kv[ct] + de) * qS[jl * 64 + ct * 16 + dl] * SCALING;
                if (mask_i && mjS[jl]) x = -3.0e38f;
                wv[r] = x;
                tm = fmaxf(tm, x);
            }
            tm = fmaxf(tm, __shfl_xor(tm, 16, 32));   // rows split across lane pair
            float mn = fmaxf(m_[ct], tm);
            float sc = __expf(m_[ct] - mn);
            float ss = s_[ct] * sc;
            float oo = o_[ct] * sc;
#pragma unroll
            for (int r = 0; r < 8; ++r) {
                int jl = r + ((lane >> 4) << 3);
                float p = __expf(wv[r] - mn);
                ss += p;
                oo += p * vS[jl * 64 + ct * 16 + dl];
            }
            s_[ct] = ss; o_[ct] = oo; m_[ct] = mn;
        }
    }

#pragma unroll
    for (int ct = 0; ct < 4; ++ct) {
        float st = s_[ct] + __shfl_xor(s_[ct], 16, 32);
        float ot = o_[ct] + __shfl_xor(o_[ct], 16, 32);
        float res = ot / st;
        if (lane < 16)
            attn_out[(size_t)(b * NTOK + i) * DIMC + d0 + ct * 16 + lane] = res;
    }
}

// ---------------------------------------------------------------------------
// LayerNorm over last dim (256); one wave per row.
// ---------------------------------------------------------------------------
__global__ __launch_bounds__(256)
void ln_kernel(const float* __restrict__ X, const float* __restrict__ g,
               const float* __restrict__ bt, float* __restrict__ Y, int rows) {
    const int lane = threadIdx.x & 31;
    const int w = threadIdx.x >> 5;
    const int row = blockIdx.x * 8 + w;
    if (row >= rows) return;
    const float* xr = X + (size_t)row * DIMC;
    float v[8];
    float sum = 0.0f;
#pragma unroll
    for (int u = 0; u < 8; ++u) { v[u] = xr[lane + 32 * u]; sum += v[u]; }
#pragma unroll
    for (int off = 16; off > 0; off >>= 1) sum += __shfl_xor(sum, off, 32);
    float mu = sum * (1.0f / DIMC);
    float var = 0.0f;
#pragma unroll
    for (int u = 0; u < 8; ++u) { float d = v[u] - mu; var += d * d; }
#pragma unroll
    for (int off = 16; off > 0; off >>= 1) var += __shfl_xor(var, off, 32);
    var *= (1.0f / DIMC);
    float inv = rsqrtf(var + 1e-5f);
#pragma unroll
    for (int u = 0; u < 8; ++u) {
        int c = lane + 32 * u;
        Y[(size_t)row * DIMC + c] = (v[u] - mu) * inv * g[c] + bt[c];
    }
}

// ---------------------------------------------------------------------------
extern "C" void kernel_launch(void* const* d_in, const int* in_sizes, int n_in,
                              void* d_out, int out_size, void* d_ws, size_t ws_size,
                              hipStream_t stream) {
    (void)in_sizes; (void)n_in; (void)out_size; (void)ws_size;
    const float* x    = (const float*)d_in[0];
    const float* pos  = (const float*)d_in[1];
    const unsigned char* mask = (const unsigned char*)d_in[2];
    const float* Wqkv = (const float*)d_in[3];
    const float* bqkv = (const float*)d_in[4];
    const float* W1   = (const float*)d_in[5];
    const float* b1   = (const float*)d_in[6];
    const float* W2   = (const float*)d_in[7];
    const float* b2   = (const float*)d_in[8];
    const float* Wo   = (const float*)d_in[9];
    const float* bo   = (const float*)d_in[10];
    const float* ln1g = (const float*)d_in[11];
    const float* ln1b = (const float*)d_in[12];
    const float* Wf1  = (const float*)d_in[13];
    const float* bf1  = (const float*)d_in[14];
    const float* Wf2  = (const float*)d_in[15];
    const float* bf2  = (const float*)d_in[16];
    const float* ln2g = (const float*)d_in[17];
    const float* ln2b = (const float*)d_in[18];

    const int M = NB * NTOK;                       // 384
    float* ws   = (float*)d_ws;
    float* qkv  = ws;                              // 384*768
    float* attn = qkv  + (size_t)M * QKVLD;        // 384*256
    float* y1   = attn + (size_t)M * DIMC;         // 384*256
    float* x1   = y1   + (size_t)M * DIMC;         // 384*256
    float* h    = x1   + (size_t)M * DIMC;         // 384*1024
    float* y2   = h    + (size_t)M * 1024;         // 384*256

    __bf16* bfbase = (__bf16*)(y2 + (size_t)M * DIMC);
    __bf16* WqkvT  = bfbase;                        // [768][256]
    __bf16* WoT    = WqkvT + (size_t)QKVLD * DIMC;  // [256][256]
    __bf16* Wf1T   = WoT   + (size_t)DIMC * DIMC;   // [1024][256]
    __bf16* Wf2T   = Wf1T  + (size_t)1024 * DIMC;   // [256][1024]
    __bf16* W2T    = Wf2T  + (size_t)DIMC * 1024;   // [256][1024]

    // 0) one-time weight bf16 convert + transpose to [N][K]
    cvt_transpose_kernel<<<(DIMC * QKVLD + 255) / 256, 256, 0, stream>>>(Wqkv, WqkvT, DIMC, QKVLD);
    cvt_transpose_kernel<<<(DIMC * DIMC  + 255) / 256, 256, 0, stream>>>(Wo,   WoT,   DIMC, DIMC);
    cvt_transpose_kernel<<<(DIMC * 1024  + 255) / 256, 256, 0, stream>>>(Wf1,  Wf1T,  DIMC, 1024);
    cvt_transpose_kernel<<<(1024 * DIMC  + 255) / 256, 256, 0, stream>>>(Wf2,  Wf2T,  1024, DIMC);
    cvt_transpose_kernel<<<(HID  * DIMC  + 255) / 256, 256, 0, stream>>>(W2,   W2T,   HID,  DIMC);

    // 1) qkv = x @ Wqkv + bqkv
    gemm_bf16_kernel<<<dim3(QKVLD / 32, M / 128), 256, 0, stream>>>(
        x, WqkvT, bqkv, nullptr, qkv, M, DIMC, QKVLD, 0);

    // 2) fused distance-attention
    attn_kernel<<<dim3(DIMC / 64, NTOK / 16, NB), 512, 0, stream>>>(
        qkv, pos, mask, W1, b1, W2T, b2, attn);

    // 3) y1 = attn @ Wo + bo + x ; x1 = LN1(y1)
    gemm_bf16_kernel<<<dim3(DIMC / 32, M / 128), 256, 0, stream>>>(
        attn, WoT, bo, x, y1, M, DIMC, DIMC, 0);
    ln_kernel<<<dim3(M / 8), 256, 0, stream>>>(y1, ln1g, ln1b, x1, M);

    // 4) h = relu(x1 @ Wf1 + bf1) ; y2 = h @ Wf2 + bf2 + x1 ; out = LN2(y2)
    gemm_bf16_kernel<<<dim3(1024 / 32, M / 128), 256, 0, stream>>>(
        x1, Wf1T, bf1, nullptr, h, M, DIMC, 1024, 1);
    gemm_bf16_kernel<<<dim3(DIMC / 32, M / 128), 256, 0, stream>>>(
        h, Wf2T, bf2, x1, y2, M, 1024, DIMC, 0);
    ln_kernel<<<dim3(M / 8), 256, 0, stream>>>(y2, ln2g, ln2b, (float*)d_out, M);
}
